// KPlexPool_8280696946974
// MI455X (gfx1250) — matrix operand order repaired and checked by reading, not compiled, not run on hardware
//
#include <hip/hip_runtime.h>
#include <hip/hip_bf16.h>
#include <math.h>

typedef __attribute__((ext_vector_type(16))) _Float16 v16h;
typedef __attribute__((ext_vector_type(8)))  float    v8f;

#define NNODE 100000
#define NEDGE 1600000
#define NFEAT 128
#define NHID  128
#define NCLS  10
#define NGR   64
#define NCLU  50000
#define HASHN (1u << 22)
#define HEMPTY 0xFFFFFFFFFFFFFFFFull

// ---------------------------------------------------------------- utilities

__global__ void zero_f32_kernel(float* __restrict__ p, long long n) {
    long long i = (long long)blockIdx.x * blockDim.x + threadIdx.x;
    if (i < n) p[i] = 0.0f;
}

__global__ void hash_init_kernel(unsigned long long* __restrict__ keys,
                                 unsigned int* __restrict__ cnts, int n) {
    int i = blockIdx.x * blockDim.x + threadIdx.x;
    if (i < n) { keys[i] = HEMPTY; cnts[i] = 0u; }
}

// Pack [Wl;Wr] (each 128x128 f32, row-major KxN) into the WMMA B lane layout:
//   Bp[ ((kc*2 + hi)*128 + n)*16 + t ] = W[K][n],  K = kc*32 + hi*16 + t
// so a lane's 16 halves for one K-chunk are 32 contiguous bytes.
__global__ void pack_weights_kernel(const float* __restrict__ Wl,
                                    const float* __restrict__ Wr,
                                    _Float16* __restrict__ Bp) {
    int i = blockIdx.x * blockDim.x + threadIdx.x;   // 0 .. 32767
    if (i >= 256 * 128) return;
    int t    = i & 15;
    int n    = (i >> 4) & 127;
    int slab = i >> 11;                              // kc*2 + hi, 0..15
    int K = (slab >> 1) * 32 + (slab & 1) * 16 + t;  // 0..255
    float v = (K < 128) ? Wl[K * 128 + n] : Wr[(K - 128) * 128 + n];
    Bp[i] = (_Float16)v;
}

// ---------------------------------------------------- edge / node scatters

__global__ void edge_scatter_kernel(const float* __restrict__ feat,
                                    const int* __restrict__ esrc,
                                    const int* __restrict__ edst,
                                    float* __restrict__ agg,
                                    float* __restrict__ cnt) {
    long long idx = (long long)blockIdx.x * blockDim.x + threadIdx.x;
    if (idx >= (long long)NEDGE * 32) return;
    int e  = (int)(idx >> 5);
    int ch = ((int)idx & 31) * 4;
    int s = esrc[e], d = edst[e];
    const float* xs = feat + (long long)s * NFEAT + ch;
    float*       ag = agg  + (long long)d * NFEAT + ch;
    atomicAdd(&ag[0], xs[0]);
    atomicAdd(&ag[1], xs[1]);
    atomicAdd(&ag[2], xs[2]);
    atomicAdd(&ag[3], xs[3]);
    if (ch == 0) atomicAdd(&cnt[d], 1.0f);
}

__global__ void node_scatter_kernel(const float* __restrict__ feat,
                                    const int* __restrict__ assign,
                                    float* __restrict__ agg,
                                    float* __restrict__ cnt) {
    long long idx = (long long)blockIdx.x * blockDim.x + threadIdx.x;
    if (idx >= (long long)NNODE * 32) return;
    int v  = (int)(idx >> 5);
    int ch = ((int)idx & 31) * 4;
    int c = assign[v];
    const float* xs = feat + (long long)v * NFEAT + ch;
    float*       ag = agg  + (long long)c * NFEAT + ch;
    atomicAdd(&ag[0], xs[0]);
    atomicAdd(&ag[1], xs[1]);
    atomicAdd(&ag[2], xs[2]);
    atomicAdd(&ag[3], xs[3]);
    if (ch == 0) atomicAdd(&cnt[c], 1.0f);
}

__global__ void mean_kernel(const float* __restrict__ agg,
                            const float* __restrict__ cnt,
                            float* __restrict__ out, int rows) {
    long long i = (long long)blockIdx.x * blockDim.x + threadIdx.x;
    if (i >= (long long)rows * NFEAT) return;
    int r = (int)(i >> 7);
    float c = cnt[r];
    out[i] = (c > 0.0f) ? agg[i] / c : 0.0f;
}

// ------------------------------------------- edge key multiplicity (unique)

__global__ void hash_insert_kernel(const int* __restrict__ esrc,
                                   const int* __restrict__ edst,
                                   const int* __restrict__ assign,
                                   unsigned long long* __restrict__ keys,
                                   unsigned int* __restrict__ kcnt,
                                   unsigned int* __restrict__ eslot) {
    int e = blockIdx.x * blockDim.x + threadIdx.x;
    if (e >= NEDGE) return;
    long long cu = assign[esrc[e]];
    long long cv = assign[edst[e]];
    unsigned long long key = (unsigned long long)(cu * (long long)NCLU + cv);
    unsigned long long h = key * 0x9E3779B97F4A7C15ull;
    unsigned int slot = (unsigned int)(h >> 32) & (HASHN - 1u);
    for (;;) {
        unsigned long long prev = atomicCAS(&keys[slot], HEMPTY, key);
        if (prev == HEMPTY || prev == key) break;
        slot = (slot + 1u) & (HASHN - 1u);
    }
    atomicAdd(&kcnt[slot], 1u);
    eslot[e] = slot;
}

__global__ void edge_weight_kernel(const unsigned int* __restrict__ kcnt,
                                   const unsigned int* __restrict__ eslot,
                                   float* __restrict__ ew) {
    int e = blockIdx.x * blockDim.x + threadIdx.x;
    if (e >= NEDGE) return;
    ew[e] = 1.0f / (float)kcnt[eslot[e]];
}

__global__ void weighted_scatter_kernel(const float* __restrict__ xc,
                                        const int* __restrict__ esrc,
                                        const int* __restrict__ edst,
                                        const int* __restrict__ assign,
                                        const float* __restrict__ ew,
                                        float* __restrict__ agg,
                                        float* __restrict__ cnt) {
    long long idx = (long long)blockIdx.x * blockDim.x + threadIdx.x;
    if (idx >= (long long)NEDGE * 32) return;
    int e  = (int)(idx >> 5);
    int ch = ((int)idx & 31) * 4;
    int cu = assign[esrc[e]];
    int cv = assign[edst[e]];
    float w = ew[e];
    const float* xs = xc  + (long long)cu * NFEAT + ch;
    float*       ag = agg + (long long)cv * NFEAT + ch;
    atomicAdd(&ag[0], w * xs[0]);
    atomicAdd(&ag[1], w * xs[1]);
    atomicAdd(&ag[2], w * xs[2]);
    atomicAdd(&ag[3], w * xs[3]);
    if (ch == 0) atomicAdd(&cnt[cv], w);
}

// ------------------------------------------- fused SAGE layer: WMMA GEMM
// h = normalize(relu([mean_agg | x] @ [Wl;Wr] + b)), one 16-row block / WG.
// 8 waves, each owns one 16x16 output tile; K=256 in 8 chunks of 32 (f16).
// B is pre-packed to the WMMA lane layout (contiguous v16h per lane/chunk)
// and hoisted into registers before the K loop.

#define LA_STRIDE 264   // 16x256 f16 A tile, padded (no LDS bank collisions)
#define LO_STRIDE 132   // 16x128 f32 out tile, padded

__global__ __launch_bounds__(256) void sage_wmma_kernel(
    const float* __restrict__ agg, const float* __restrict__ cnt,
    const float* __restrict__ xin, const _Float16* __restrict__ Bp,
    const float* __restrict__ bias, float* __restrict__ out) {
    __shared__ _Float16 lA[16 * LA_STRIDE];
    __shared__ float    lO[16 * LO_STRIDE];
    __shared__ float    lN[16];

    const int t = threadIdx.x;
    const int row0 = blockIdx.x * 16;

    const int wave = t >> 5;
    const int lane = t & 31;
    const int m16  = lane & 15;
    const int hi   = lane >> 4;            // K half select within chunk
    const int ncol = wave * 16 + m16;      // output column

    // hoist all B chunks into registers: one coalesced 32B load per chunk
    v16h breg[8];
#pragma unroll
    for (int kc = 0; kc < 8; ++kc)
        breg[kc] = *(const v16h*)&Bp[(((kc * 2) + hi) * 128 + ncol) * 16];

    // stage A = [mean_agg | x] as f16 (16 rows x 256 K)
    for (int i = t; i < 16 * 256; i += 256) {
        int r = i >> 8;
        int k = i & 255;
        int grow = row0 + r;
        float v;
        if (k < 128) {
            float c = cnt[grow];
            v = (c > 0.0f) ? agg[(long long)grow * NFEAT + k] / c : 0.0f;
        } else {
            v = xin[(long long)grow * NFEAT + (k - 128)];
        }
        lA[r * LA_STRIDE + k] = (_Float16)v;
    }
    __syncthreads();

    v8f acc = {};
#pragma unroll
    for (int kc = 0; kc < 8; ++kc) {
        // A layout (16-bit 16x32): lanes hi=0 hold K {0..15}+base,
        // hi=1 hold K {16..31}+base of row m16 (matches packed B)
        const _Float16* ap = &lA[m16 * LA_STRIDE + kc * 32 + hi * 16];
        v16h a;
#pragma unroll
        for (int j = 0; j < 8; ++j) { a[j] = ap[j]; a[j + 8] = ap[j + 8]; }
        acc = __builtin_amdgcn_wmma_f32_16x16x32_f16(
            false, a, false, breg[kc], (short)0, acc, false, false);
    }

    // bias + relu into LDS C-tile (lane holds col ncol, rows r+8*hi)
    float bb = bias[ncol];
#pragma unroll
    for (int r = 0; r < 8; ++r) {
        float v = acc[r] + bb;
        v = v > 0.0f ? v : 0.0f;
        lO[(r + hi * 8) * LO_STRIDE + ncol] = v;
    }
    __syncthreads();

    if (t < 16) {
        float s = 0.0f;
        for (int j = 0; j < NHID; ++j) {
            float v = lO[t * LO_STRIDE + j];
            s += v * v;
        }
        lN[t] = sqrtf(s);
    }
    __syncthreads();

    for (int i = t; i < 16 * NHID; i += 256) {
        int r = i >> 7, c = i & 127;
        out[(long long)(row0 + r) * NHID + c] = lO[r * LO_STRIDE + c] / lN[r];
    }
}

// ------------------------------- cluster output layer (128 -> 10) + pooling

__global__ __launch_bounds__(256) void cluster_out_kernel(
    const float* __restrict__ aggc, const float* __restrict__ cntw,
    const float* __restrict__ xc,   const float* __restrict__ Wl,
    const float* __restrict__ Wr,   const float* __restrict__ bo,
    const int* __restrict__ bpool,  float* __restrict__ pooled,
    float* __restrict__ pooledCnt) {
    int wid  = (blockIdx.x * blockDim.x + threadIdx.x) >> 5;
    int lane = threadIdx.x & 31;
    if (wid >= NCLU) return;
    float cw = cntw[wid];
    float acc[NCLS];
#pragma unroll
    for (int c = 0; c < NCLS; ++c) acc[c] = 0.0f;
    for (int k = lane; k < NHID; k += 32) {
        float mv = (cw > 0.0f) ? aggc[(long long)wid * NHID + k] / cw : 0.0f;
        float xv = xc[(long long)wid * NHID + k];
#pragma unroll
        for (int c = 0; c < NCLS; ++c)
            acc[c] += mv * Wl[k * NCLS + c] + xv * Wr[k * NCLS + c];
    }
#pragma unroll
    for (int c = 0; c < NCLS; ++c)
        for (int off = 16; off; off >>= 1)
            acc[c] += __shfl_down(acc[c], off, 32);
    if (lane == 0) {
        float y[NCLS], s = 0.0f;
#pragma unroll
        for (int c = 0; c < NCLS; ++c) { y[c] = acc[c] + bo[c]; s += y[c] * y[c]; }
        float nrm = sqrtf(s);
        int g = bpool[wid];
#pragma unroll
        for (int c = 0; c < NCLS; ++c) atomicAdd(&pooled[g * NCLS + c], y[c] / nrm);
        atomicAdd(&pooledCnt[g], 1.0f);
    }
}

__global__ void finalize_kernel(const float* __restrict__ pooled,
                                const float* __restrict__ pooledCnt,
                                float* __restrict__ out) {
    int g = threadIdx.x;
    if (g >= NGR) return;
    float c = pooledCnt[g];
    float y[NCLS];
#pragma unroll
    for (int i = 0; i < NCLS; ++i)
        y[i] = (c > 0.0f) ? pooled[g * NCLS + i] / c : 0.0f;
    float mx = y[0];
#pragma unroll
    for (int i = 1; i < NCLS; ++i) mx = fmaxf(mx, y[i]);
    float se = 0.0f;
#pragma unroll
    for (int i = 0; i < NCLS; ++i) se += __expf(y[i] - mx);
    float lse = mx + __logf(se);
#pragma unroll
    for (int i = 0; i < NCLS; ++i) out[g * NCLS + i] = y[i] - lse;
}

// ---------------------------------------------------------------- launcher

static inline size_t alignup(size_t x) { return (x + 255) & ~(size_t)255; }

extern "C" void kernel_launch(void* const* d_in, const int* in_sizes, int n_in,
                              void* d_out, int out_size, void* d_ws, size_t ws_size,
                              hipStream_t stream) {
    const float* x      = (const float*)d_in[0];
    const float* Wl_in  = (const float*)d_in[1];
    const float* Wr_in  = (const float*)d_in[2];
    const float* b_in   = (const float*)d_in[3];
    const float* Wl_h   = (const float*)d_in[4];
    const float* Wr_h   = (const float*)d_in[5];
    const float* b_h    = (const float*)d_in[6];
    const float* Wl_out = (const float*)d_in[7];
    const float* Wr_out = (const float*)d_in[8];
    const float* b_out  = (const float*)d_in[9];
    const int*   esrc   = (const int*)d_in[10];
    const int*   edst   = (const int*)d_in[11];
    const int*   clus   = (const int*)d_in[12];
    const int*   bpool  = (const int*)d_in[13];
    float* out = (float*)d_out;

    // workspace carve-out
    char* base = (char*)d_ws;
    size_t off = 0;
    float* bufA = (float*)(base + off); off = alignup(off + sizeof(float) * (size_t)NNODE * NFEAT);
    float* h1   = (float*)(base + off); off = alignup(off + sizeof(float) * (size_t)NNODE * NFEAT);
    float* h2   = (float*)(base + off); off = alignup(off + sizeof(float) * (size_t)NNODE * NFEAT);
    float* cnt  = (float*)(base + off); off = alignup(off + sizeof(float) * (size_t)NNODE);
    _Float16* B1 = (_Float16*)(base + off); off = alignup(off + sizeof(_Float16) * 256 * 128);
    _Float16* B2 = (_Float16*)(base + off); off = alignup(off + sizeof(_Float16) * 256 * 128);
    unsigned long long* hkeys = (unsigned long long*)(base + off);
    off = alignup(off + sizeof(unsigned long long) * (size_t)HASHN);
    unsigned int* hcnt  = (unsigned int*)(base + off); off = alignup(off + sizeof(unsigned int) * (size_t)HASHN);
    unsigned int* eslot = (unsigned int*)(base + off); off = alignup(off + sizeof(unsigned int) * (size_t)NEDGE);
    float* ew = (float*)(base + off); off = alignup(off + sizeof(float) * (size_t)NEDGE);
    float* pooled    = (float*)(base + off); off = alignup(off + sizeof(float) * NGR * NCLS);
    float* pooledCnt = (float*)(base + off); off = alignup(off + sizeof(float) * NGR);
    float* xc = h1;  // reuse h1 for cluster features after layer 2

    const int TB = 256;
    const long long nfeatAll = (long long)NNODE * NFEAT;
    dim3 zgridN((unsigned)((nfeatAll + TB - 1) / TB));
    dim3 zgridNc((unsigned)((NNODE + TB - 1) / TB));
    dim3 egrid((unsigned)(((long long)NEDGE * 32 + TB - 1) / TB));
    dim3 ngrid((unsigned)(((long long)NNODE * 32 + TB - 1) / TB));
    dim3 ggrid(NNODE / 16);       // 6250 blocks, 16 rows each
    dim3 cgridF((unsigned)(((long long)NCLU * NFEAT + TB - 1) / TB));
    dim3 cgridW((NCLU * 32 + TB - 1) / TB); // wave per cluster row
    dim3 hgrid((HASHN + TB - 1) / TB);
    dim3 egr1((NEDGE + TB - 1) / TB);

    // weight packs (f32 -> f16, B lane layout, K contiguous per lane half)
    pack_weights_kernel<<<128, TB, 0, stream>>>(Wl_in, Wr_in, B1);
    pack_weights_kernel<<<128, TB, 0, stream>>>(Wl_h, Wr_h, B2);

    // ---- layer 1
    zero_f32_kernel<<<zgridN, TB, 0, stream>>>(bufA, nfeatAll);
    zero_f32_kernel<<<zgridNc, TB, 0, stream>>>(cnt, NNODE);
    edge_scatter_kernel<<<egrid, TB, 0, stream>>>(x, esrc, edst, bufA, cnt);
    sage_wmma_kernel<<<ggrid, TB, 0, stream>>>(bufA, cnt, x, B1, b_in, h1);

    // ---- layer 2
    zero_f32_kernel<<<zgridN, TB, 0, stream>>>(bufA, nfeatAll);
    zero_f32_kernel<<<zgridNc, TB, 0, stream>>>(cnt, NNODE);
    edge_scatter_kernel<<<egrid, TB, 0, stream>>>(h1, esrc, edst, bufA, cnt);
    sage_wmma_kernel<<<ggrid, TB, 0, stream>>>(bufA, cnt, h1, B2, b_h, h2);

    // ---- cluster pooling: xc = segment_mean(h2, cluster_assign, C)
    zero_f32_kernel<<<zgridN, TB, 0, stream>>>(bufA, nfeatAll);
    zero_f32_kernel<<<zgridNc, TB, 0, stream>>>(cnt, NNODE);
    node_scatter_kernel<<<ngrid, TB, 0, stream>>>(h2, clus, bufA, cnt);
    mean_kernel<<<cgridF, TB, 0, stream>>>(bufA, cnt, xc, NCLU);

    // ---- edge multiplicity weights (unique of (cu,cv))
    hash_init_kernel<<<hgrid, TB, 0, stream>>>(hkeys, hcnt, (int)HASHN);
    hash_insert_kernel<<<egr1, TB, 0, stream>>>(esrc, edst, clus, hkeys, hcnt, eslot);
    edge_weight_kernel<<<egr1, TB, 0, stream>>>(hcnt, eslot, ew);

    // ---- weighted cluster SAGE + pooled mean + log_softmax
    zero_f32_kernel<<<cgridF, TB, 0, stream>>>(bufA, (long long)NCLU * NFEAT);
    zero_f32_kernel<<<zgridNc, TB, 0, stream>>>(cnt, NCLU);
    zero_f32_kernel<<<1, TB, 0, stream>>>(pooled, NGR * NCLS);
    zero_f32_kernel<<<1, TB, 0, stream>>>(pooledCnt, NGR);
    weighted_scatter_kernel<<<egrid, TB, 0, stream>>>(xc, esrc, edst, clus, ew, bufA, cnt);
    cluster_out_kernel<<<cgridW, TB, 0, stream>>>(bufA, cnt, xc, Wl_out, Wr_out,
                                                  b_out, bpool, pooled, pooledCnt);
    finalize_kernel<<<1, 64, 0, stream>>>(pooled, pooledCnt, out);
    (void)n_in; (void)in_sizes; (void)out_size; (void)ws_size;
}
